// QKVProj_24378234372647
// MI455X (gfx1250) — matrix operand order repaired and checked by reading, not compiled, not run on hardware
//
#include <hip/hip_runtime.h>
#include <hip/hip_bf16.h>

// QKV fused projection as one GEMM:
//   M = 4*2048 = 8192, K = 4096, N = 3*4096 = 12288
// fp32 inputs converted to bf16 on the LDS-fill path, v_wmma_f32_16x16x32_bf16
// with fp32 accumulation, fp32 output.

#define D_MODEL 4096
#define N_OUT   12288

#define BM 128
#define BN 128
#define BK 32
#define LDS_STRIDE 40   // bf16 elements per row (32 + 8 pad) -> 80B row stride

typedef __attribute__((ext_vector_type(16))) __bf16 v16bf;
typedef __attribute__((ext_vector_type(8)))  __bf16 v8bf;
typedef __attribute__((ext_vector_type(4)))  __bf16 v4bf;
typedef __attribute__((ext_vector_type(8)))  float  v8f;
typedef __attribute__((ext_vector_type(4)))  float  v4f;

union AFrag { v16bf v; v8bf h[2]; };

__global__ __launch_bounds__(256)
void qkv_gemm_bf16_wmma(const float* __restrict__ x,
                        const float* __restrict__ Wq,
                        const float* __restrict__ Wk,
                        const float* __restrict__ Wv,
                        float* __restrict__ out)
{
    __shared__ __bf16 ldsA[2][BM * LDS_STRIDE];
    __shared__ __bf16 ldsB[2][BN * LDS_STRIDE];

    const int ntile = blockIdx.x;          // 0..95  (output-column tile)
    const int mtile = blockIdx.y;          // 0..63  (row tile)

    // Column tiles 0..31 -> Wq, 32..63 -> Wk, 64..95 -> Wv (concatenated N dim)
    const float* W = (ntile < 32) ? Wq : ((ntile < 64) ? Wk : Wv);
    const int w_row0 = (ntile & 31) * BN;  // row offset within selected weight
    const int m_row0 = mtile * BM;

    const int tid    = threadIdx.x;
    const int lane   = tid & 31;
    const int wid    = tid >> 5;           // 0..7
    const int wave_m = wid & 1;            // 2 tiles of 64 rows
    const int wave_n = wid >> 1;           // 4 tiles of 32 cols

    // Global->LDS fill mapping: each thread owns a half-row (16 fp32) of both tiles
    const int ld_row = tid >> 1;           // 0..127
    const int ld_col = (tid & 1) * 16;     // 0 or 16

    const float* gA = x + (size_t)(m_row0 + ld_row) * D_MODEL + ld_col;
    const float* gB = W + (size_t)(w_row0 + ld_row) * D_MODEL + ld_col;
    const int lds_off = ld_row * LDS_STRIDE + ld_col;

    v8f acc[4][2] = {};

    const int KSTEPS = D_MODEL / BK;       // 128

    v4f ra[4], rb[4];                      // staged global data (b128 loads)

    auto load_global = [&](int k) {
        const float* pa = gA + k * BK;
        const float* pb = gB + k * BK;
        #pragma unroll
        for (int i = 0; i < 4; ++i) {
            ra[i] = *(const v4f*)(pa + 4 * i);
            rb[i] = *(const v4f*)(pb + 4 * i);
        }
    };

    auto store_lds = [&](int buf) {
        #pragma unroll
        for (int i = 0; i < 4; ++i) {
            v4bf a, b;
            #pragma unroll
            for (int c = 0; c < 4; ++c) {
                a[c] = (__bf16)ra[i][c];   // v_cvt f32 -> bf16 (RNE)
                b[c] = (__bf16)rb[i][c];
            }
            *(v4bf*)(&ldsA[buf][lds_off + 4 * i]) = a;   // ds_store_b64
            *(v4bf*)(&ldsB[buf][lds_off + 4 * i]) = b;
        }
    };

    auto compute = [&](int buf) {
        // B fragments (shared across the 4 A tiles):
        //   lane n%16 = column N, K chunk 0..15 (lanes 0-15) / 16..31 (lanes 16-31),
        //   contiguous in LDS row.
        AFrag bfrag[2];
        const int bofs = (lane >> 4) * 16;
        #pragma unroll
        for (int u = 0; u < 2; ++u) {
            const int brow = wave_n * 32 + u * 16 + (lane & 15);
            const __bf16* p = &ldsB[buf][brow * LDS_STRIDE + bofs];
            bfrag[u].h[0] = *(const v8bf*)(p);       // ds_load_b128
            bfrag[u].h[1] = *(const v8bf*)(p + 8);
        }
        // A fragments: lane m%16 = row M, K chunk 0..15 (lanes 0-15) / 16..31 (lanes 16-31)
        const int kofs = (lane >> 4) * 16;
        #pragma unroll
        for (int t = 0; t < 4; ++t) {
            const int arow = wave_m * 64 + t * 16 + (lane & 15);
            const __bf16* p = &ldsA[buf][arow * LDS_STRIDE + kofs];
            AFrag afrag;
            afrag.h[0] = *(const v8bf*)(p);          // ds_load_b128
            afrag.h[1] = *(const v8bf*)(p + 8);
            #pragma unroll
            for (int u = 0; u < 2; ++u) {
                acc[t][u] = __builtin_amdgcn_wmma_f32_16x16x32_bf16(
                    /*neg_a=*/false, afrag.v,
                    /*neg_b=*/false, bfrag[u].v,
                    /*c_mod=*/(short)0, acc[t][u],
                    /*reuse_a=*/false, /*reuse_b=*/false);
            }
        }
    };

    // Prologue: fill buffer 0
    load_global(0);
    store_lds(0);
    __syncthreads();

    // Pipelined main loop: global loads for k+1 in flight under the WMMAs on k,
    // then convert+store into the alternate LDS buffer.
    for (int k = 0; k < KSTEPS; ++k) {
        const int buf = k & 1;
        const bool more = (k + 1) < KSTEPS;
        if (more) load_global(k + 1);
        compute(buf);
        if (more) store_lds(buf ^ 1);
        __syncthreads();
    }

    // Write out: C/D layout -> lane n = col, VGPR v = row (+8 for lanes 16-31)
    #pragma unroll
    for (int t = 0; t < 4; ++t) {
        const int row = m_row0 + wave_m * 64 + t * 16 + (lane >> 4) * 8;
        #pragma unroll
        for (int u = 0; u < 2; ++u) {
            const int col = ntile * BN + wave_n * 32 + u * 16 + (lane & 15);
            float* p = out + (size_t)row * N_OUT + col;
            #pragma unroll
            for (int v = 0; v < 8; ++v)
                p[(size_t)v * N_OUT] = acc[t][u][v];
        }
    }
}

extern "C" void kernel_launch(void* const* d_in, const int* in_sizes, int n_in,
                              void* d_out, int out_size, void* d_ws, size_t ws_size,
                              hipStream_t stream) {
    (void)in_sizes; (void)n_in; (void)out_size; (void)d_ws; (void)ws_size;
    const float* x  = (const float*)d_in[0];
    const float* Wq = (const float*)d_in[1];
    const float* Wk = (const float*)d_in[2];
    const float* Wv = (const float*)d_in[3];
    float* out = (float*)d_out;

    dim3 grid(N_OUT / BN, (4 * 2048) / BM);   // (96, 64)
    qkv_gemm_bf16_wmma<<<grid, 256, 0, stream>>>(x, Wq, Wk, Wv, out);
}
